// Recovery_74698071212402
// MI455X (gfx1250) — matrix operand order repaired and checked by reading, not compiled, not run on hardware
//
#include <hip/hip_runtime.h>
#include <hip/hip_bf16.h>

#define B_   128
#define T_   256
#define H_   512
#define G3_  1536
#define F_   64
#define L_   3

#define NWG  32
#define TPB  256   // 8 waves/block, 256 waves total = 8 m-tiles x 32 c-tiles

typedef __bf16 bf16;
typedef __attribute__((ext_vector_type(16))) __bf16 v16bf;
typedef __attribute__((ext_vector_type(8)))  __bf16 v8bf;
typedef __attribute__((ext_vector_type(8)))  float  v8f;

__device__ __forceinline__ float sigmoidf_(float x) {
  return 1.0f / (1.0f + __expf(-x));
}

// ---- WMMA operand loaders (wave32; contiguous k per lane half, same order for A and B) ----
// A: 16x32 bf16, lane<16 -> M=lane, K=k0..k0+15 contiguous;
//    lane>=16 -> M=lane-16, K=k0+16..k0+31 contiguous.
__device__ __forceinline__ v16bf load_A(const bf16* __restrict__ base, long ld,
                                        int m0, int k0, int lane) {
  int m  = m0 + (lane & 15);
  int kk = k0 + ((lane & 16) ? 16 : 0);
  const bf16* p = base + (long)m * ld + kk;
  union { v16bf v; v8bf h[2]; } u;
  u.h[0] = *(const v8bf*)(p);        // 16B
  u.h[1] = *(const v8bf*)(p + 8);    // 16B
  return u.v;
}

// B: 32x16 bf16 from pre-transposed weight Wt[N][K] (row-major, ld=K):
// lane<16 -> N=lane, K=k0..k0+15 contiguous; lane>=16 -> N=lane-16, K=k0+16..k0+31.
__device__ __forceinline__ v16bf load_B(const bf16* __restrict__ baseT, long ld,
                                        int n0, int k0, int lane) {
  int n  = n0 + (lane & 15);
  int kk = k0 + ((lane & 16) ? 16 : 0);
  return *(const v16bf*)(baseT + (long)n * ld + kk);   // 32B
}

__device__ __forceinline__ v8f wmma_bf16(v16bf a, v16bf b, v8f c) {
  return __builtin_amdgcn_wmma_f32_16x16x32_bf16(
      false, a, false, b, (short)0, c, false, false);
}

// ---- persistent-grid barrier (device-scope atomics, sense via generation) ----
__device__ __forceinline__ void grid_barrier(unsigned* bar) {
  __syncthreads();
  if (threadIdx.x == 0) {
    unsigned* cnt = bar;
    unsigned* gen = bar + 1;
    unsigned g = __hip_atomic_load(gen, __ATOMIC_RELAXED, __HIP_MEMORY_SCOPE_AGENT);
    unsigned a = __hip_atomic_fetch_add(cnt, 1u, __ATOMIC_ACQ_REL, __HIP_MEMORY_SCOPE_AGENT);
    if (a == NWG - 1) {
      __hip_atomic_store(cnt, 0u, __ATOMIC_RELAXED, __HIP_MEMORY_SCOPE_AGENT);
      __hip_atomic_fetch_add(gen, 1u, __ATOMIC_RELEASE, __HIP_MEMORY_SCOPE_AGENT);
    } else {
      while (__hip_atomic_load(gen, __ATOMIC_ACQUIRE, __HIP_MEMORY_SCOPE_AGENT) == g) {
        __builtin_amdgcn_s_sleep(2);
      }
    }
  }
  __syncthreads();
}

__device__ __forceinline__ float* hfb(float* hf, int l, int par) {
  return hf + (long)(l * 2 + par) * (B_ * H_);
}
__device__ __forceinline__ bf16* hbb(bf16* hb, int l, int par) {
  return hb + (long)(l * 2 + par) * (B_ * H_);
}

// One wave computes h_new for its 16x16 tile (m0, c0): six 16x16x512 gate
// sub-GEMMs (x@W, h@U for z/r/h columns) + local GRU gate math.
__device__ __forceinline__ void layer_phase(
    const bf16* __restrict__ Xbf, long x_ld,
    const bf16* __restrict__ Hprev_bf,
    const float* __restrict__ Hprev_f,
    const bf16* __restrict__ Wt, const bf16* __restrict__ Ut,   // [G3_, H_]
    const float* __restrict__ bi, const float* __restrict__ br, // [G3_]
    float* __restrict__ Hout_f, bf16* __restrict__ Hout_bf,
    int m0, int c0, int lane)
{
  v8f axz = {}, axr = {}, axh = {}, ahz = {}, ahr = {}, ahh = {};
  for (int k = 0; k < H_; k += 32) {
    v16bf ax  = load_A(Xbf,      x_ld, m0, k, lane);
    v16bf ah  = load_A(Hprev_bf, H_,   m0, k, lane);
    v16bf bwz = load_B(Wt, H_, c0,           k, lane);
    v16bf bwr = load_B(Wt, H_, c0 + H_,      k, lane);
    v16bf bwh = load_B(Wt, H_, c0 + 2 * H_,  k, lane);
    v16bf buz = load_B(Ut, H_, c0,           k, lane);
    v16bf bur = load_B(Ut, H_, c0 + H_,      k, lane);
    v16bf buh = load_B(Ut, H_, c0 + 2 * H_,  k, lane);
    axz = wmma_bf16(ax, bwz, axz);
    axr = wmma_bf16(ax, bwr, axr);
    axh = wmma_bf16(ax, bwh, axh);
    ahz = wmma_bf16(ah, buz, ahz);
    ahr = wmma_bf16(ah, bur, ahr);
    ahh = wmma_bf16(ah, buh, ahh);
  }
  // C/D layout: lane<16 -> N=lane, rows m0+i (i=0..7); lane>=16 -> N=lane-16, rows m0+8+i
  int c  = c0 + (lane & 15);
  int mb = m0 + ((lane & 16) ? 8 : 0);
  float biz = bi[c], bir = bi[c + H_], bih = bi[c + 2 * H_];
  float brz = br[c], brr = br[c + H_], brh = br[c + 2 * H_];
#pragma unroll
  for (int i = 0; i < 8; ++i) {
    int m = mb + i;
    float hp = Hprev_f[(long)m * H_ + c];
    float z  = sigmoidf_(axz[i] + ahz[i] + biz + brz);
    float r  = sigmoidf_(axr[i] + ahr[i] + bir + brr);
    float hc = tanhf(axh[i] + bih + r * (ahh[i] + brh));
    float hn = z * hp + (1.0f - z) * hc;
    Hout_f[(long)m * H_ + c]  = hn;
    Hout_bf[(long)m * H_ + c] = (bf16)hn;
  }
}

__device__ __forceinline__ void dense_phase(
    const bf16* __restrict__ H2bf, const bf16* __restrict__ WdT,
    const float* __restrict__ bd, float* __restrict__ out,
    int t, int m0, int n0, int lane)
{
  v8f acc = {};
  for (int k = 0; k < H_; k += 32) {
    v16bf a = load_A(H2bf, H_, m0, k, lane);
    v16bf b = load_B(WdT,  H_, n0, k, lane);
    acc = wmma_bf16(a, b, acc);
  }
  int n  = n0 + (lane & 15);
  int mb = m0 + ((lane & 16) ? 8 : 0);
  float bdv = bd[n];
#pragma unroll
  for (int i = 0; i < 8; ++i) {
    int m = mb + i;
    out[(long)m * (T_ * F_) + (long)t * F_ + n] = sigmoidf_(acc[i] + bdv);
  }
}

__global__ void __launch_bounds__(TPB, 1) gru3_persistent(
    const float* __restrict__ b_in, const float* __restrict__ b_rec,
    const float* __restrict__ bd,
    const bf16* __restrict__ Hbf, const bf16* __restrict__ WtT,
    const bf16* __restrict__ UtT, const bf16* __restrict__ WdT,
    float* __restrict__ hf, bf16* __restrict__ hb,
    float* __restrict__ out, unsigned* __restrict__ bar)
{
  int lane = threadIdx.x & 31;
  int wid  = blockIdx.x * (TPB / 32) + (threadIdx.x >> 5);
  int m0 = (wid >> 5) * 16;   // 8 m-tiles (batch)
  int c0 = (wid & 31) * 16;   // 32 c-tiles (hidden)
  int dm0 = (wid >> 2) * 16;  // dense tiles for waves 0..31
  int dn0 = (wid & 3) * 16;

  for (int t = 0; t < T_; ++t) {
    int p = t & 1, q = p ^ 1;   // read parity p, write parity q

    // layer 0 (input from Hbf[:, t, :]); waves 0..31 also emit dense(t-1)
    layer_phase(Hbf + (long)t * H_, (long)T_ * H_,
                hbb(hb, 0, p), hfb(hf, 0, p),
                WtT, UtT, b_in, b_rec,
                hfb(hf, 0, q), hbb(hb, 0, q), m0, c0, lane);
    if (t > 0 && wid < 32)
      dense_phase(hbb(hb, 2, p), WdT, bd, out, t - 1, dm0, dn0, lane);
    grid_barrier(bar);

    // layer 1 (input = h0(t) in parity q)
    layer_phase(hbb(hb, 0, q), H_, hbb(hb, 1, p), hfb(hf, 1, p),
                WtT + 1L * G3_ * H_, UtT + 1L * G3_ * H_,
                b_in + G3_, b_rec + G3_,
                hfb(hf, 1, q), hbb(hb, 1, q), m0, c0, lane);
    grid_barrier(bar);

    // layer 2
    layer_phase(hbb(hb, 1, q), H_, hbb(hb, 2, p), hfb(hf, 2, p),
                WtT + 2L * G3_ * H_, UtT + 2L * G3_ * H_,
                b_in + 2 * G3_, b_rec + 2 * G3_,
                hfb(hf, 2, q), hbb(hb, 2, q), m0, c0, lane);
    grid_barrier(bar);
  }
  // final dense for t = T_-1 (h2 written at t=255 into parity (256&1)=0)
  if (wid < 32)
    dense_phase(hbb(hb, 2, 0), WdT, bd, out, T_ - 1, dm0, dn0, lane);
}

// ---- prep: transpose+convert weights to bf16 [N][K] ----
__global__ void prep_weights(const float* __restrict__ Ws, const float* __restrict__ Us,
                             const float* __restrict__ Wd,
                             bf16* __restrict__ WtT, bf16* __restrict__ UtT,
                             bf16* __restrict__ WdT)
{
  long stride = (long)gridDim.x * blockDim.x;
  long i0 = (long)blockIdx.x * blockDim.x + threadIdx.x;
  const long nW = (long)L_ * G3_ * H_;
  for (long idx = i0; idx < nW; idx += stride) {
    long l = idx / ((long)G3_ * H_);
    long rem = idx - l * ((long)G3_ * H_);
    long n = rem / H_, k = rem - n * H_;
    long src = l * (long)H_ * G3_ + k * G3_ + n;
    WtT[idx] = (bf16)Ws[src];
    UtT[idx] = (bf16)Us[src];
  }
  const long nD = (long)F_ * H_;
  for (long idx = i0; idx < nD; idx += stride) {
    long n = idx / H_, k = idx - n * H_;
    WdT[idx] = (bf16)Wd[k * F_ + n];
  }
}

// ---- prep: H -> bf16, zero h-state parity 0, reset barrier ----
__global__ void prep_state(const float* __restrict__ Hin, bf16* __restrict__ Hbf,
                           float* __restrict__ hf, bf16* __restrict__ hb,
                           unsigned* __restrict__ bar)
{
  long stride = (long)gridDim.x * blockDim.x;
  long i0 = (long)blockIdx.x * blockDim.x + threadIdx.x;
  if (i0 == 0) { bar[0] = 0u; bar[1] = 0u; }
  const long nH = (long)B_ * T_ * H_;
  for (long idx = i0; idx < nH; idx += stride)
    Hbf[idx] = (bf16)Hin[idx];
  const long nS = (long)L_ * B_ * H_;
  for (long idx = i0; idx < nS; idx += stride) {
    long l = idx / ((long)B_ * H_);
    long r = idx - l * ((long)B_ * H_);
    long o = (l * 2) * (long)(B_ * H_) + r;   // parity 0
    hf[o] = 0.0f;
    hb[o] = (bf16)0.0f;
  }
}

extern "C" void kernel_launch(void* const* d_in, const int* in_sizes, int n_in,
                              void* d_out, int out_size, void* d_ws, size_t ws_size,
                              hipStream_t stream) {
  const float* Hin   = (const float*)d_in[0];
  const float* Ws    = (const float*)d_in[1];
  const float* Us    = (const float*)d_in[2];
  const float* b_in  = (const float*)d_in[3];
  const float* b_rec = (const float*)d_in[4];
  const float* Wd    = (const float*)d_in[5];
  const float* bd    = (const float*)d_in[6];
  float* out = (float*)d_out;

  char* w = (char*)d_ws;
  auto carve = [&](size_t bytes) -> char* {
    char* p = w;
    w += (bytes + 255) & ~(size_t)255;
    return p;
  };
  unsigned* bar = (unsigned*)carve(256);
  bf16*  WtT = (bf16*)carve((size_t)L_ * G3_ * H_ * sizeof(bf16));
  bf16*  UtT = (bf16*)carve((size_t)L_ * G3_ * H_ * sizeof(bf16));
  bf16*  WdT = (bf16*)carve((size_t)F_ * H_ * sizeof(bf16));
  bf16*  Hbf = (bf16*)carve((size_t)B_ * T_ * H_ * sizeof(bf16));
  float* hf  = (float*)carve((size_t)L_ * 2 * B_ * H_ * sizeof(float));
  bf16*  hb  = (bf16*)carve((size_t)L_ * 2 * B_ * H_ * sizeof(bf16));

  prep_weights<<<2048, 256, 0, stream>>>(Ws, Us, Wd, WtT, UtT, WdT);
  prep_state<<<4096, 256, 0, stream>>>(Hin, Hbf, hf, hb, bar);
  gru3_persistent<<<NWG, TPB, 0, stream>>>(b_in, b_rec, bd,
                                           Hbf, WtT, UtT, WdT,
                                           hf, hb, out, bar);
}
